// AttentionBlock_89824946029065
// MI455X (gfx1250) — compile-verified, matrix operand order checked by reading, not run on hardware
//
#include <hip/hip_runtime.h>
#include <math.h>

typedef __attribute__((ext_vector_type(16))) _Float16 v16h;
typedef __attribute__((ext_vector_type(8)))  _Float16 v8h;
typedef __attribute__((ext_vector_type(8)))  float    v8f;

#define NTOK   50176
#define EPS    1e-5f
#define MASKF  -30000.0f

// ---------------- WMMA fragment helpers (CDNA5 16x16x32 f16 layouts) ----------------
// A matrix 16x32 (MxK) from row-major f16 [..][ld]:
//   lane L: M = m0 + (L&15), K base = k0 + 8*(L>>4)
//   elements 0..7 -> K = kb+0..7 ; elements 8..15 -> K = kb+16..23
// Also serves as B-fragment for X @ W^T with row-major W[N][K], and as the
// dots B-fragment from the row-major K slab [token][d].
__device__ __forceinline__ v16h frag_rowmajor(const _Float16* base, int ld, int m0, int k0) {
  int lane = threadIdx.x & 31;
  const _Float16* p = base + (m0 + (lane & 15)) * ld + k0 + ((lane >> 4) << 3);
  v16h r;
#pragma unroll
  for (int e = 0; e < 8; ++e) r[e] = p[e];
#pragma unroll
  for (int e = 0; e < 8; ++e) r[8 + e] = p[16 + e];
  return r;
}

// B matrix 32x16 (KxN) from row-major f16 [K][ld] (K contiguous rows).
__device__ __forceinline__ v16h frag_kxn(const _Float16* base, int ld, int k0, int n0) {
  int lane = threadIdx.x & 31;
  const _Float16* p = base + (k0 + ((lane >> 4) << 3)) * ld + n0 + (lane & 15);
  v16h r;
#pragma unroll
  for (int e = 0; e < 8; ++e) r[e] = p[e * ld];
#pragma unroll
  for (int e = 0; e < 8; ++e) r[8 + e] = p[(16 + e) * ld];
  return r;
}

__device__ __forceinline__ v8f wmma16(v16h a, v16h b, v8f c) {
  return __builtin_amdgcn_wmma_f32_16x16x32_f16(false, a, false, b, (short)0, c, false, false);
}

// LayerNorm over 96 cols, half-wave (16 lanes) per row, 64 rows (4 passes),
// rows >= nvalid zero-filled (WMMA padding).
__device__ __forceinline__ void ln_rows_to_f16(const float* src, _Float16* dst,
                                               const float* g, const float* bb,
                                               int nvalid, int tid) {
  int wv = tid >> 5, lane = tid & 31, l = lane & 15;
#pragma unroll 1
  for (int pass = 0; pass < 4; ++pass) {
    int row = pass * 16 + wv * 2 + (lane >> 4);
    if (row < nvalid) {
      const float* xr = src + row * 96;
      float s = 0.f;
#pragma unroll
      for (int e = 0; e < 6; ++e) s += xr[l + e * 16];
      s += __shfl_xor(s, 1); s += __shfl_xor(s, 2); s += __shfl_xor(s, 4); s += __shfl_xor(s, 8);
      float mean = s * (1.0f / 96.0f);
      float vv = 0.f;
#pragma unroll
      for (int e = 0; e < 6; ++e) { float d = xr[l + e * 16] - mean; vv += d * d; }
      vv += __shfl_xor(vv, 1); vv += __shfl_xor(vv, 2); vv += __shfl_xor(vv, 4); vv += __shfl_xor(vv, 8);
      float rs = rsqrtf(vv * (1.0f / 96.0f) + EPS);
#pragma unroll
      for (int e = 0; e < 6; ++e) {
        int c = l + e * 16;
        dst[row * 96 + c] = (_Float16)((xr[c] - mean) * rs * g[c] + bb[c]);
      }
    } else if (row < 64) {
#pragma unroll
      for (int e = 0; e < 6; ++e) dst[row * 96 + l + e * 16] = (_Float16)0.0f;
    }
  }
}

// =====================================================================================
// Kernel 1: windowed attention. 1 block = 4 windows, 256 threads = 8 wave32.
// =====================================================================================
__global__ __launch_bounds__(256) void win_attn_kernel(
    const float* __restrict__ x, const float* __restrict__ w_qkv,
    const float* __restrict__ w_out, const float* __restrict__ b_out,
    const float* __restrict__ rel_bias, const float* __restrict__ ln1_g,
    const float* __restrict__ ln1_b, float* __restrict__ y) {
  __shared__ _Float16 sWq[384 * 96];                  // w_qkv  [N=384][K=96] f16
  __shared__ _Float16 sWo[96 * 128];                  // w_out  [N=96][K=128] f16
  __shared__ __align__(16) _Float16 sBiasT[4 * 64 * 64];  // masked (bias*scale)^T [h][n][m]
  __shared__ float    sBout[96], sG[96], sB[96];
  __shared__ float    sX[49 * 96];                    // window input f32 (residual)
  __shared__ _Float16 sH[64 * 96];                    // LN1 output f16, rows>=49 zero
  __shared__ _Float16 sQKV[64 * 384];                 // [m][0:128)=Q |[128:256)=K |[256:384)=V
  __shared__ _Float16 sAtt[4 * 64 * 64];
  __shared__ _Float16 sO[64 * 128];

  const int tid = threadIdx.x, wv = tid >> 5, lane = tid & 31;
  const float scale = 0.17677669529663689f;  // 32^-0.5

  for (int i = tid; i < 384 * 96; i += 256) sWq[i] = (_Float16)w_qkv[i];
  for (int i = tid; i < 96 * 128; i += 256) sWo[i] = (_Float16)w_out[i];
  for (int i = tid; i < 4 * 64 * 64; i += 256) {
    int h = i >> 12, n = (i >> 6) & 63, m = i & 63;
    float v = MASKF;
    if (m < 49 && n < 49) {
      int r0 = m / 7 - n / 7 + 6, r1 = m % 7 - n % 7 + 6;
      v = rel_bias[(r0 * 13 + r1) * 4 + h] * scale;
    }
    sBiasT[i] = (_Float16)v;
  }
  if (tid < 96) { sBout[tid] = b_out[tid]; sG[tid] = ln1_g[tid]; sB[tid] = ln1_b[tid]; }
  __syncthreads();

#pragma unroll 1
  for (int wloop = 0; wloop < 4; ++wloop) {
    const int wid = blockIdx.x * 4 + wloop;
    const int b = wid >> 10, rem = wid & 1023;
    const int rx = rem >> 5, ry = rem & 31;

    // ---- load window (49 rows of 96 contiguous floats) ----
    for (int i = tid; i < 49 * 96; i += 256) {
      int m = i / 96, c = i - m * 96;
      int wi = m / 7, wj = m - wi * 7;
      int ntok = (rx * 7 + wi) * 224 + (ry * 7 + wj);
      sX[i] = x[((size_t)b * NTOK + ntok) * 96 + c];
    }
    // prefetch next window's rows (3 x 128B lines per 384B row)
    if (wloop < 3 && tid < 49) {
      int wid2 = wid + 1;
      int b2 = wid2 >> 10, rem2 = wid2 & 1023, rx2 = rem2 >> 5, ry2 = rem2 & 31;
      int wi = tid / 7, wj = tid - wi * 7;
      const float* row = &x[((size_t)b2 * NTOK + (rx2 * 7 + wi) * 224 + (ry2 * 7 + wj)) * 96];
      __builtin_prefetch(row, 0, 1);
      __builtin_prefetch(row + 32, 0, 1);
      __builtin_prefetch(row + 64, 0, 1);
    }
    __syncthreads();

    ln_rows_to_f16(sX, sH, sG, sB, 49, tid);
    __syncthreads();

    // ---- QKV: (64x96)@(96x384) -> sQKV. Outer N-tile caches B-fragments;
    //      inner M loop NOT unrolled to bound the live register set. ----
#pragma unroll 1
    for (int j = 0; j < 3; ++j) {
      int nt = wv + j * 8;
      v16h b0 = frag_rowmajor(sWq, 96, nt * 16, 0);
      v16h b1 = frag_rowmajor(sWq, 96, nt * 16, 32);
      v16h b2 = frag_rowmajor(sWq, 96, nt * 16, 64);
      int ncol = nt * 16 + (lane & 15);
#pragma unroll 1
      for (int mt = 0; mt < 4; ++mt) {
        v8f acc = {};
        acc = wmma16(frag_rowmajor(sH, 96, mt * 16, 0),  b0, acc);
        acc = wmma16(frag_rowmajor(sH, 96, mt * 16, 32), b1, acc);
        acc = wmma16(frag_rowmajor(sH, 96, mt * 16, 64), b2, acc);
#pragma unroll
        for (int r = 0; r < 8; ++r) {
          int m = mt * 16 + r + ((lane >> 4) << 3);
          sQKV[m * 384 + ncol] = (_Float16)acc[r];
        }
      }
    }
    __syncthreads();

    // ---- dots + bias + softmax: wave = (head, M-half); K^T B-fragments hoisted ----
    {
      const int hh = wv >> 1, mh = wv & 1;
      v16h bK[4];
#pragma unroll
      for (int nt = 0; nt < 4; ++nt)   // B[d][j] = K[j][d]: A-style frag over K slab
        bK[nt] = frag_rowmajor(sQKV + 128 + hh * 32, 384, nt * 16, 0);
#pragma unroll 1
      for (int mi = 0; mi < 2; ++mi) {
        int mt = mh * 2 + mi;
        v16h a = frag_rowmajor(sQKV, 384, mt * 16, hh * 32);  // Q slice
        v8f acc[4];
#pragma unroll
        for (int nt = 0; nt < 4; ++nt) { v8f z = {}; acc[nt] = wmma16(a, bK[nt], z); }

        int nl = lane & 15, mb = mt * 16 + ((lane >> 4) << 3);
#pragma unroll
        for (int nt = 0; nt < 4; ++nt) {
          int n = nt * 16 + nl;
          v8h bv = *(const v8h*)(sBiasT + ((hh * 64 + n) << 6) + mb);  // 8 contig m's
#pragma unroll
          for (int r = 0; r < 8; ++r)
            acc[nt][r] = fmaf(acc[nt][r], scale, (float)bv[r]);       // masked = -30000
        }
#pragma unroll
        for (int r = 0; r < 8; ++r) {
          float mx = fmaxf(fmaxf(acc[0][r], acc[1][r]), fmaxf(acc[2][r], acc[3][r]));
          mx = fmaxf(mx, __shfl_xor(mx, 1)); mx = fmaxf(mx, __shfl_xor(mx, 2));
          mx = fmaxf(mx, __shfl_xor(mx, 4)); mx = fmaxf(mx, __shfl_xor(mx, 8));
          float p[4], sum = 0.f;
#pragma unroll
          for (int nt = 0; nt < 4; ++nt) { p[nt] = __expf(acc[nt][r] - mx); sum += p[nt]; }
          sum += __shfl_xor(sum, 1); sum += __shfl_xor(sum, 2);
          sum += __shfl_xor(sum, 4); sum += __shfl_xor(sum, 8);
          float inv = 1.0f / sum;
          int m = mb + r;
#pragma unroll
          for (int nt = 0; nt < 4; ++nt)
            sAtt[(hh * 64 + m) * 64 + nt * 16 + nl] = (_Float16)(p[nt] * inv);
        }
      }
    }
    __syncthreads();

    // ---- O = attn @ V : V B-fragments hoisted, A shared across N-tiles ----
    {
      const int hh = wv >> 1, mh = wv & 1;
      v16h bV[4];  // [kc][nt]
#pragma unroll
      for (int kc = 0; kc < 2; ++kc)
#pragma unroll
        for (int nt = 0; nt < 2; ++nt)
          bV[kc * 2 + nt] = frag_kxn(sQKV + 256 + hh * 32, 384, kc * 32, nt * 16);
#pragma unroll 1
      for (int mi = 0; mi < 2; ++mi) {
        int mt = mh * 2 + mi;
        v16h a0 = frag_rowmajor(sAtt + hh * 4096, 64, mt * 16, 0);
        v16h a1 = frag_rowmajor(sAtt + hh * 4096, 64, mt * 16, 32);
#pragma unroll
        for (int nt = 0; nt < 2; ++nt) {
          v8f acc = {};
          acc = wmma16(a0, bV[nt], acc);
          acc = wmma16(a1, bV[2 + nt], acc);
          int n = nt * 16 + (lane & 15);
#pragma unroll
          for (int r = 0; r < 8; ++r) {
            int m = mt * 16 + r + ((lane >> 4) << 3);
            sO[m * 128 + hh * 32 + n] = (_Float16)acc[r];
          }
        }
      }
    }
    __syncthreads();

    // ---- out proj: fixed M-tile/wave, 3 accumulators share each A-fragment ----
    {
      const int mt = wv >> 1, nt0 = (wv & 1) * 3;
      v8f acc[3] = {};
#pragma unroll 1
      for (int kc = 0; kc < 4; ++kc) {
        v16h a = frag_rowmajor(sO, 128, mt * 16, kc * 32);
        acc[0] = wmma16(a, frag_rowmajor(sWo, 128, (nt0 + 0) * 16, kc * 32), acc[0]);
        acc[1] = wmma16(a, frag_rowmajor(sWo, 128, (nt0 + 1) * 16, kc * 32), acc[1]);
        acc[2] = wmma16(a, frag_rowmajor(sWo, 128, (nt0 + 2) * 16, kc * 32), acc[2]);
      }
#pragma unroll
      for (int j = 0; j < 3; ++j) {
        int n = (nt0 + j) * 16 + (lane & 15);
#pragma unroll
        for (int r = 0; r < 8; ++r) {
          int m = mt * 16 + r + ((lane >> 4) << 3);
          if (m < 49) {
            float vvl = acc[j][r] + sBout[n] + sX[m * 96 + n];
            int wi = m / 7, wj = m % 7;
            int ntok = (rx * 7 + wi) * 224 + (ry * 7 + wj);
            y[((size_t)b * NTOK + ntok) * 96 + n] = vvl;
          }
        }
      }
    }
    __syncthreads();
  }
}

// =====================================================================================
// Kernel 2: LN2 + FFN(GELU) + residual, in-place on y. 1 block = 256 tokens (4x64).
// =====================================================================================
__global__ __launch_bounds__(256) void ffn_kernel(
    const float* __restrict__ ln2_g, const float* __restrict__ ln2_b,
    const float* __restrict__ w1, const float* __restrict__ b1,
    const float* __restrict__ w2, const float* __restrict__ b2,
    float* __restrict__ y) {
  __shared__ _Float16 sW1[384 * 96];   // [N=384][K=96]
  __shared__ _Float16 sW2[96 * 384];   // [N=96][K=384]
  __shared__ float    sB1[384], sB2[96], sG[96], sBt[96];
  __shared__ float    sY[64 * 96];
  __shared__ _Float16 sH[64 * 96];
  __shared__ _Float16 sAct[64 * 384];

  const int tid = threadIdx.x, wv = tid >> 5, lane = tid & 31;
  for (int i = tid; i < 384 * 96; i += 256) sW1[i] = (_Float16)w1[i];
  for (int i = tid; i < 96 * 384; i += 256) sW2[i] = (_Float16)w2[i];
  for (int i = tid; i < 384; i += 256) sB1[i] = b1[i];
  if (tid < 96) { sB2[tid] = b2[tid]; sG[tid] = ln2_g[tid]; sBt[tid] = ln2_b[tid]; }
  __syncthreads();

#pragma unroll 1
  for (int chunk = 0; chunk < 4; ++chunk) {
    const size_t tok0 = ((size_t)blockIdx.x * 4 + chunk) * 64;
    for (int i = tid; i < 64 * 96; i += 256) sY[i] = y[tok0 * 96 + i];
    __syncthreads();

    ln_rows_to_f16(sY, sH, sG, sBt, 64, tid);
    __syncthreads();

    // fc1: outer N-tile caches B-fragments; inner M loop not unrolled
#pragma unroll 1
    for (int j = 0; j < 3; ++j) {
      int nt = wv + j * 8;
      v16h b0 = frag_rowmajor(sW1, 96, nt * 16, 0);
      v16h b1f = frag_rowmajor(sW1, 96, nt * 16, 32);
      v16h b2f = frag_rowmajor(sW1, 96, nt * 16, 64);
      int n = nt * 16 + (lane & 15);
      float bias = sB1[n];
#pragma unroll 1
      for (int mt = 0; mt < 4; ++mt) {
        v8f acc = {};
        acc = wmma16(frag_rowmajor(sH, 96, mt * 16, 0),  b0,  acc);
        acc = wmma16(frag_rowmajor(sH, 96, mt * 16, 32), b1f, acc);
        acc = wmma16(frag_rowmajor(sH, 96, mt * 16, 64), b2f, acc);
#pragma unroll
        for (int r = 0; r < 8; ++r) {
          int m = mt * 16 + r + ((lane >> 4) << 3);
          float xv = acc[r] + bias;
          float g = 0.5f * xv * (1.0f + erff(xv * 0.70710678118654752f));
          sAct[m * 384 + n] = (_Float16)g;
        }
      }
    }
    __syncthreads();

    // fc2: fixed M-tile/wave, 3 accumulators share each A-fragment; + residual
    {
      const int mt = wv >> 1, nt0 = (wv & 1) * 3;
      v8f acc[3] = {};
#pragma unroll 1
      for (int kc = 0; kc < 12; ++kc) {
        v16h a = frag_rowmajor(sAct, 384, mt * 16, kc * 32);
        acc[0] = wmma16(a, frag_rowmajor(sW2, 384, (nt0 + 0) * 16, kc * 32), acc[0]);
        acc[1] = wmma16(a, frag_rowmajor(sW2, 384, (nt0 + 1) * 16, kc * 32), acc[1]);
        acc[2] = wmma16(a, frag_rowmajor(sW2, 384, (nt0 + 2) * 16, kc * 32), acc[2]);
      }
#pragma unroll
      for (int j = 0; j < 3; ++j) {
        int n = (nt0 + j) * 16 + (lane & 15);
#pragma unroll
        for (int r = 0; r < 8; ++r) {
          int m = mt * 16 + r + ((lane >> 4) << 3);
          y[(tok0 + m) * 96 + n] = acc[j][r] + sB2[n] + sY[m * 96 + n];
        }
      }
    }
    __syncthreads();
  }
}

extern "C" void kernel_launch(void* const* d_in, const int* in_sizes, int n_in,
                              void* d_out, int out_size, void* d_ws, size_t ws_size,
                              hipStream_t stream) {
  (void)in_sizes; (void)n_in; (void)d_ws; (void)ws_size; (void)out_size;
  const float* x        = (const float*)d_in[0];
  const float* w_qkv    = (const float*)d_in[1];
  const float* w_out    = (const float*)d_in[2];
  const float* b_out    = (const float*)d_in[3];
  const float* rel_bias = (const float*)d_in[4];
  const float* ln1_g    = (const float*)d_in[5];
  const float* ln1_b    = (const float*)d_in[6];
  const float* ln2_g    = (const float*)d_in[7];
  const float* ln2_b    = (const float*)d_in[8];
  const float* w1       = (const float*)d_in[9];
  const float* b1       = (const float*)d_in[10];
  const float* w2       = (const float*)d_in[11];
  const float* b2       = (const float*)d_in[12];
  float* y = (float*)d_out;

  // 8192 windows / 4 per block
  hipLaunchKernelGGL(win_attn_kernel, dim3(2048), dim3(256), 0, stream,
                     x, w_qkv, w_out, b_out, rel_bias, ln1_g, ln1_b, y);
  // 401408 tokens / 256 per block
  hipLaunchKernelGGL(ffn_kernel, dim3(1568), dim3(256), 0, stream,
                     ln2_g, ln2_b, w1, b1, w2, b2, y);
}
